// MHA_3770981285896
// MI455X (gfx1250) — matrix operand order repaired and checked by reading, not compiled, not run on hardware
//
#include <hip/hip_runtime.h>
#include <math.h>
#include <stdint.h>

// ---------------------------------------------------------------------------
// MI455X (gfx1250, wave32) MHA block:
//   LN -> (Q,K,V) proj [WMMA f16] -> flash attention [WMMA f16] ->
//   head-major merge (free, layout identity) -> out proj + residual [WMMA f16]
// GEMM tiles staged with double-buffered GLOBAL_LOAD_ASYNC_TO_LDS_B128.
// ---------------------------------------------------------------------------

typedef __attribute__((ext_vector_type(16))) _Float16 v16h;
typedef __attribute__((ext_vector_type(8)))  _Float16 v8h;
typedef __attribute__((ext_vector_type(8)))  float    v8f;
typedef __attribute__((ext_vector_type(4)))  int      v4i;

typedef __attribute__((address_space(1))) v4i as1_v4i;
typedef __attribute__((address_space(3))) v4i as3_v4i;

#define D_MODEL 1024
#define N_TOK   2048
#define BS      2
#define NHEAD   16
#define DHEAD   64
#define BT      (BS * N_TOK)   // 4096 rows

// ---- gfx1250 async global->LDS copy (ASYNCcnt path), compile-safe gating ----
#if defined(__has_builtin)
#  if __has_builtin(__builtin_amdgcn_global_load_async_to_lds_b128)
#    define USE_ASYNC_LDS 1
#  endif
#  if __has_builtin(__builtin_amdgcn_s_wait_asynccnt)
#    define HAVE_WAIT_ASYNC 1
#  endif
#endif

__device__ __forceinline__ void async_ld16(void* lds, const void* g) {
#if USE_ASYNC_LDS
    __builtin_amdgcn_global_load_async_to_lds_b128(
        (as1_v4i*)(uintptr_t)g,
        (as3_v4i*)(uint32_t)(uintptr_t)lds,
        0, 0);
#else
    (void)lds; (void)g;
#endif
}
__device__ __forceinline__ void wait_async_le4() {
#if USE_ASYNC_LDS
#  if HAVE_WAIT_ASYNC
    __builtin_amdgcn_s_wait_asynccnt(4);
#  else
    asm volatile("s_wait_asynccnt 4" ::: "memory");
#  endif
#endif
}
__device__ __forceinline__ void wait_async0() {
#if USE_ASYNC_LDS
#  if HAVE_WAIT_ASYNC
    __builtin_amdgcn_s_wait_asynccnt(0);
#  else
    asm volatile("s_wait_asynccnt 0" ::: "memory");
#  endif
#endif
}

// ---------------------------------------------------------------------------
// Kernel 1: LayerNorm over d_model, emit f16 h
// ---------------------------------------------------------------------------
__global__ __launch_bounds__(256) void ln_kernel(const float* __restrict__ q,
                                                 const float* __restrict__ gamma,
                                                 const float* __restrict__ beta,
                                                 _Float16* __restrict__ h16) {
    int row = blockIdx.x;
    int tid = threadIdx.x;
    const float* x = q + (size_t)row * D_MODEL;
    float4 v = ((const float4*)x)[tid];           // 256 threads * 4 = 1024
    float s  = v.x + v.y + v.z + v.w;
    float s2 = v.x * v.x + v.y * v.y + v.z * v.z + v.w * v.w;

    __shared__ float rs[256], rq[256];
    rs[tid] = s; rq[tid] = s2;
    __syncthreads();
    for (int off = 128; off > 0; off >>= 1) {
        if (tid < off) { rs[tid] += rs[tid + off]; rq[tid] += rq[tid + off]; }
        __syncthreads();
    }
    float mu  = rs[0] * (1.0f / D_MODEL);
    float var = rq[0] * (1.0f / D_MODEL) - mu * mu;
    float rstd = rsqrtf(var + 1e-5f);

    int base = tid * 4;
    float xs[4] = {v.x, v.y, v.z, v.w};
    _Float16* o = h16 + (size_t)row * D_MODEL + base;
#pragma unroll
    for (int j = 0; j < 4; ++j) {
        float hv = (xs[j] - mu) * rstd * gamma[base + j] + beta[base + j];
        o[j] = (_Float16)hv;
    }
}

// ---------------------------------------------------------------------------
// Kernel 2: transpose-convert weight fp32 [K][N] -> f16 [N][K]
// ---------------------------------------------------------------------------
__global__ __launch_bounds__(256) void wconv_kernel(const float* __restrict__ W,
                                                    _Float16* __restrict__ Wt) {
    __shared__ float tile[32][33];
    int n0 = blockIdx.x * 32, k0 = blockIdx.y * 32;
    int tx = threadIdx.x, ty = threadIdx.y;            // 32 x 8
#pragma unroll
    for (int i = 0; i < 32; i += 8)
        tile[ty + i][tx] = W[(size_t)(k0 + ty + i) * D_MODEL + n0 + tx];
    __syncthreads();
#pragma unroll
    for (int i = 0; i < 32; i += 8)
        Wt[(size_t)(n0 + ty + i) * D_MODEL + k0 + tx] = (_Float16)tile[tx][ty + i];
}

// ---------------------------------------------------------------------------
// Kernel 3: WMMA GEMM  C[M=4096,N=1024] = A[M,1024] * Bt[N,1024]^T
//   MODE 0: +bias, scatter to head layout [B,H,T,64] f16      (Q, K proj)
//   MODE 1: +bias, relu, scatter to transposed [B,H,64,T] f16 (V proj)
//   MODE 2: +bias +residual(q), fp32 out [M,N]                (out proj)
// Block tile 128x128, 8 waves of 64x32, K-step 32.
// Double-buffered async tile staging: tile i+1 DMAs while tile i computes.
// ---------------------------------------------------------------------------
template <int MODE>
__global__ __launch_bounds__(256) void gemm_kernel(const _Float16* __restrict__ A,
                                                   const _Float16* __restrict__ Bt,
                                                   const float* __restrict__ bias,
                                                   const float* __restrict__ resid,
                                                   void* __restrict__ out) {
    const int K = D_MODEL;
    int tid = threadIdx.x;
    int wid = tid >> 5, lane = tid & 31;
    int tileM = blockIdx.y * 128, tileN = blockIdx.x * 128;
    int wm = (wid >> 2) * 64, wn = (wid & 3) * 32;
    int mrow = lane & 15, hi = lane >> 4;

    v8f acc[4][2] = {};

    int lrow = tid >> 1;             // 0..127
    int lk   = (tid & 1) * 16;       // 0 or 16

    const _Float16* agBase = A  + (size_t)(tileM + lrow) * K + lk;
    const _Float16* bgBase = Bt + (size_t)(tileN + lrow) * K + lk;

#if USE_ASYNC_LDS
    __shared__ __align__(16) _Float16 As[2][128 * 40];  // stride 40 halves: conflict-free
    __shared__ __align__(16) _Float16 Bs[2][128 * 40];

    // prologue: DMA tile 0 into buffer 0
    {
        _Float16* as = &As[0][lrow * 40 + lk];
        _Float16* bs = &Bs[0][lrow * 40 + lk];
        async_ld16(as,     agBase);
        async_ld16(as + 8, agBase + 8);
        async_ld16(bs,     bgBase);
        async_ld16(bs + 8, bgBase + 8);
    }

    for (int k0 = 0; k0 < K; k0 += 32) {
        int cur = (k0 >> 5) & 1;
        if (k0 + 32 < K) {
            // issue tile i+1 into the other buffer, leave it in flight
            _Float16* as = &As[cur ^ 1][lrow * 40 + lk];
            _Float16* bs = &Bs[cur ^ 1][lrow * 40 + lk];
            const _Float16* ag = agBase + k0 + 32;
            const _Float16* bg = bgBase + k0 + 32;
            async_ld16(as,     ag);
            async_ld16(as + 8, ag + 8);
            async_ld16(bs,     bg);
            async_ld16(bs + 8, bg + 8);
            wait_async_le4();                    // tile i landed; i+1 still in flight
        } else {
            wait_async0();
        }
        __syncthreads();                         // everyone's tile-i writes visible
#else
    __shared__ __align__(16) _Float16 As[1][128 * 40];
    __shared__ __align__(16) _Float16 Bs[1][128 * 40];

    for (int k0 = 0; k0 < K; k0 += 32) {
        int cur = 0;
        const _Float16* ag = agBase + k0;
        const _Float16* bg = bgBase + k0;
        if (k0 + 32 < K) {                       // gfx1250 global_prefetch_b8, near
            __builtin_prefetch(ag + 32, 0, 3);
            __builtin_prefetch(bg + 32, 0, 3);
        }
        v8h a0 = *(const v8h*)ag;
        v8h a1 = *(const v8h*)(ag + 8);
        v8h b0 = *(const v8h*)bg;
        v8h b1 = *(const v8h*)(bg + 8);
        __syncthreads();
        _Float16* as = &As[0][lrow * 40 + lk];
        _Float16* bs = &Bs[0][lrow * 40 + lk];
        *(v8h*)as       = a0;
        *(v8h*)(as + 8) = a1;
        *(v8h*)bs       = b0;
        *(v8h*)(bs + 8) = b1;
        __syncthreads();
#endif

        // A fragments: lane = M row; K elems split {0..7,16..23} / {8..15,24..31}
        v16h af[4];
#pragma unroll
        for (int ms = 0; ms < 4; ++ms) {
            const _Float16* p = &As[cur][(wm + ms * 16 + mrow) * 40 + hi * 8];
            v16h t;
            *(v8h*)&t       = *(const v8h*)p;
            *((v8h*)&t + 1) = *(const v8h*)(p + 16);
            af[ms] = t;
        }
        // B fragments: lane = N col; K elems split {0..7,16..23} / {8..15,24..31}
        v16h bf[2];
#pragma unroll
        for (int ns = 0; ns < 2; ++ns) {
            const _Float16* p = &Bs[cur][(wn + ns * 16 + mrow) * 40 + hi * 8];
            v16h t;
            *(v8h*)&t       = *(const v8h*)p;
            *((v8h*)&t + 1) = *(const v8h*)(p + 16);
            bf[ns] = t;
        }
#pragma unroll
        for (int ms = 0; ms < 4; ++ms)
#pragma unroll
            for (int ns = 0; ns < 2; ++ns)
                acc[ms][ns] = __builtin_amdgcn_wmma_f32_16x16x32_f16(
                    false, af[ms], false, bf[ns], (short)0, acc[ms][ns], false, false);

        __syncthreads();   // all reads done before this buffer is re-DMA'd next iter
    }

    // epilogue: C element r -> M = r + 8*hi, N = lane&15
    // fully unrolled: constant vector indices, no scratch spills
#pragma unroll
    for (int ms = 0; ms < 4; ++ms)
#pragma unroll
        for (int ns = 0; ns < 2; ++ns)
#pragma unroll
            for (int r = 0; r < 8; ++r) {
                int mg = tileM + wm + ms * 16 + r + hi * 8;
                int ng = tileN + wn + ns * 16 + mrow;
                float vv = acc[ms][ns][r] + bias[ng];
                if (MODE == 2) {
                    size_t idx = (size_t)mg * D_MODEL + ng;
                    ((float*)out)[idx] = vv + resid[idx];
                } else {
                    if (MODE == 1) vv = fmaxf(vv, 0.0f);
                    int b = mg >> 11, t = mg & 2047;     // row -> (batch, token)
                    int hd = ng >> 6, d = ng & 63;       // col -> (head, dim)
                    size_t idx = (MODE == 0)
                        ? ((size_t)((b * NHEAD + hd) * N_TOK + t) * DHEAD + d)
                        : ((size_t)((b * NHEAD + hd) * DHEAD + d) * N_TOK + t);
                    ((_Float16*)out)[idx] = (_Float16)vv;
                }
            }
}

// ---------------------------------------------------------------------------
// Kernel 4: flash attention per (b,h).  qh,kh: [BH, T, 64], vt: [BH, 64, T].
// Each wave owns 16 query rows; keys streamed in chunks of 32.
// ---------------------------------------------------------------------------
__global__ __launch_bounds__(256) void attn_kernel(const _Float16* __restrict__ qh,
                                                   const _Float16* __restrict__ kh,
                                                   const _Float16* __restrict__ vt,
                                                   _Float16* __restrict__ o16) {
    __shared__ __align__(16) _Float16 pbuf[8][16 * 32];  // per-wave P staging

    int wid = threadIdx.x >> 5, lane = threadIdx.x & 31;
    int mrow = lane & 15, hi = lane >> 4;
    int bh = blockIdx.x;
    int q0 = blockIdx.y * 128 + wid * 16;

    const _Float16* Q  = qh  + (size_t)bh * N_TOK * DHEAD;
    const _Float16* Kp = kh  + (size_t)bh * N_TOK * DHEAD;
    const _Float16* Vp = vt  + (size_t)bh * DHEAD * N_TOK;
    _Float16*       Op = o16 + (size_t)bh * N_TOK * DHEAD;
    _Float16*       pl = pbuf[wid];

    // Q fragments (A layout), K-dim = d: two frags covering d in [0,32),[32,64)
    v16h aq[2];
#pragma unroll
    for (int f = 0; f < 2; ++f) {
        const _Float16* p = Q + (size_t)(q0 + mrow) * DHEAD + f * 32 + hi * 8;
        v16h t;
        *(v8h*)&t       = *(const v8h*)p;
        *((v8h*)&t + 1) = *(const v8h*)(p + 16);
        aq[f] = t;
    }

    v8f acc[4] = {};
    float mrun[8], lrun[8];
#pragma unroll
    for (int r = 0; r < 8; ++r) { mrun[r] = -INFINITY; lrun[r] = 0.0f; }

    for (int key0 = 0; key0 < N_TOK; key0 += 32) {
        // ---- S = Q K^T for 16q x 32k : 2 tiles x 2 K-steps ----
        v8f s[2] = {};
#pragma unroll
        for (int t = 0; t < 2; ++t)
#pragma unroll
            for (int p = 0; p < 2; ++p) {
                const _Float16* kp = Kp + (size_t)(key0 + t * 16 + mrow) * DHEAD
                                        + p * 32 + hi * 8;
                v16h bk;
                *(v8h*)&bk       = *(const v8h*)kp;
                *((v8h*)&bk + 1) = *(const v8h*)(kp + 16);
                s[t] = __builtin_amdgcn_wmma_f32_16x16x32_f16(
                    false, aq[p], false, bk, (short)0, s[t], false, false);
            }

        // ---- online softmax (rows live across 16-lane halves, wave32) ----
        float p0[8], p1[8];
#pragma unroll
        for (int r = 0; r < 8; ++r) {
            float a = s[0][r] * 0.125f;            // 1/sqrt(64)
            float b = s[1][r] * 0.125f;
            float mx = fmaxf(a, b);
#pragma unroll
            for (int off = 1; off < 16; off <<= 1)
                mx = fmaxf(mx, __shfl_xor(mx, off, 32));
            float mnew  = fmaxf(mrun[r], mx);
            float scale = __expf(mrun[r] - mnew);
            float e0 = __expf(a - mnew);
            float e1 = __expf(b - mnew);
            float rsum = e0 + e1;
#pragma unroll
            for (int off = 1; off < 16; off <<= 1)
                rsum += __shfl_xor(rsum, off, 32);
            lrun[r] = lrun[r] * scale + rsum;
            mrun[r] = mnew;
#pragma unroll
            for (int dn = 0; dn < 4; ++dn) acc[dn][r] *= scale;
            p0[r] = e0; p1[r] = e1;
        }

        // ---- stage P (C layout) -> LDS -> reload as A fragment ----
#pragma unroll
        for (int r = 0; r < 8; ++r) {
            int row = r + hi * 8;
            pl[row * 32 + mrow]      = (_Float16)p0[r];
            pl[row * 32 + 16 + mrow] = (_Float16)p1[r];
        }
        asm volatile("" ::: "memory");   // keep DS store->load order (same wave, in-order)
        v16h ap;
        {
            const _Float16* p = pl + mrow * 32 + hi * 8;
            *(v8h*)&ap       = *(const v8h*)p;
            *((v8h*)&ap + 1) = *(const v8h*)(p + 16);
        }

        // ---- O += P @ V : 4 tiles along d (V stored transposed [64,T]) ----
#pragma unroll
        for (int dn = 0; dn < 4; ++dn) {
            const _Float16* vp = Vp + (size_t)(dn * 16 + mrow) * N_TOK + key0 + hi * 8;
            v16h bv;
            *(v8h*)&bv       = *(const v8h*)vp;
            *((v8h*)&bv + 1) = *(const v8h*)(vp + 16);
            acc[dn] = __builtin_amdgcn_wmma_f32_16x16x32_f16(
                false, ap, false, bv, (short)0, acc[dn], false, false);
        }
    }

    // ---- normalize + store O in [B,H,T,64] f16 (== torch merged layout) ----
#pragma unroll
    for (int r = 0; r < 8; ++r) {
        float inv = 1.0f / lrun[r];
        int t = q0 + r + hi * 8;
#pragma unroll
        for (int dn = 0; dn < 4; ++dn)
            Op[(size_t)t * DHEAD + dn * 16 + mrow] = (_Float16)(acc[dn][r] * inv);
    }
}

// ---------------------------------------------------------------------------
// Host launcher
// ---------------------------------------------------------------------------
extern "C" void kernel_launch(void* const* d_in, const int* in_sizes, int n_in,
                              void* d_out, int out_size, void* d_ws, size_t ws_size,
                              hipStream_t stream) {
    (void)in_sizes; (void)n_in; (void)out_size; (void)ws_size;

    const float* q     = (const float*)d_in[0];
    // d_in[1] (k) and d_in[2] (v) are ignored by the reference
    const float* gamma = (const float*)d_in[3];
    const float* beta  = (const float*)d_in[4];
    const float* Wq    = (const float*)d_in[5];
    const float* bq    = (const float*)d_in[6];
    const float* Wk    = (const float*)d_in[7];
    const float* bk    = (const float*)d_in[8];
    const float* Wv    = (const float*)d_in[9];
    const float* bv    = (const float*)d_in[10];
    const float* Wo    = (const float*)d_in[11];
    const float* bo    = (const float*)d_in[12];

    char* w = (char*)d_ws;
    const size_t MB = 1024 * 1024;
    _Float16* h16   = (_Float16*)(w + 0);        //  8 MB  [4096][1024]
    _Float16* Wq16t = (_Float16*)(w + 8  * MB);  //  2 MB  [N][K]
    _Float16* Wk16t = (_Float16*)(w + 10 * MB);
    _Float16* Wv16t = (_Float16*)(w + 12 * MB);
    _Float16* Wo16t = (_Float16*)(w + 14 * MB);
    _Float16* qhh   = (_Float16*)(w + 16 * MB);  //  8 MB  [B,H,T,64]
    _Float16* khh   = (_Float16*)(w + 24 * MB);  //  8 MB  [B,H,T,64]
    _Float16* vtt   = (_Float16*)(w + 32 * MB);  //  8 MB  [B,H,64,T]
    _Float16* o16   = (_Float16*)(w + 40 * MB);  //  8 MB  [B,H,T,64]

    // 1. LayerNorm -> f16
    ln_kernel<<<BT, 256, 0, stream>>>(q, gamma, beta, h16);

    // 2. weights fp32 -> f16 transposed
    dim3 wg(32, 32), wb(32, 8);
    wconv_kernel<<<wg, wb, 0, stream>>>(Wq, Wq16t);
    wconv_kernel<<<wg, wb, 0, stream>>>(Wk, Wk16t);
    wconv_kernel<<<wg, wb, 0, stream>>>(Wv, Wv16t);
    wconv_kernel<<<wg, wb, 0, stream>>>(Wo, Wo16t);

    // 3. projections (WMMA GEMM): grid = (N/128, M/128)
    dim3 gg(D_MODEL / 128, BT / 128);
    gemm_kernel<0><<<gg, 256, 0, stream>>>(h16, Wq16t, bq, nullptr, qhh);
    gemm_kernel<0><<<gg, 256, 0, stream>>>(h16, Wk16t, bk, nullptr, khh);
    gemm_kernel<1><<<gg, 256, 0, stream>>>(h16, Wv16t, bv, nullptr, vtt);

    // 4. attention: 32 (b,h) pairs x 16 query blocks of 128
    dim3 ag(BS * NHEAD, N_TOK / 128);
    attn_kernel<<<ag, 256, 0, stream>>>(qhh, khh, vtt, o16);

    // 5. out proj + bias + residual (merged == o16 flat, torch-faithful)
    gemm_kernel<2><<<gg, 256, 0, stream>>>(o16, Wo16t, bo, q, d_out);
}
